// DGLRGCN_57492432224404
// MI455X (gfx1250) — matrix operand order repaired and checked by reading, not compiled, not run on hardware
//
#include <hip/hip_runtime.h>
#include <math.h>

typedef __bf16 bf16_t;
typedef __bf16 v8bf  __attribute__((ext_vector_type(8)));
typedef __bf16 v16bf __attribute__((ext_vector_type(16)));
typedef float  v8f   __attribute__((ext_vector_type(8)));

// ---------------------------------------------------------------------------
// Zero fill
__global__ void k_zero(float* __restrict__ p, long long n) {
    long long i = (long long)blockIdx.x * blockDim.x + threadIdx.x;
    if (i < n) p[i] = 0.f;
}

// ---------------------------------------------------------------------------
// Weight transpose + f32->bf16: dst[n*ldd + koff + k] = src[k*Nsrc + n]
// (K fixed at 256 per sub-block; rows n >= Nsrc are zero-padded)
__global__ void k_wtrans(const float* __restrict__ src, int Nsrc,
                         bf16_t* __restrict__ dst, int ldd, int koff, int Ndst) {
    int i = blockIdx.x * blockDim.x + threadIdx.x;
    if (i >= Ndst * 256) return;
    int n = i >> 8, k = i & 255;
    float v = (n < Nsrc) ? src[(size_t)k * Nsrc + n] : 0.f;
    dst[(size_t)n * ldd + (size_t)koff + k] = (bf16_t)v;
}

// ---------------------------------------------------------------------------
// f32 [M,256] -> bf16 column slice of a [M,ldd] matrix
__global__ void k_cvt_cols(const float* __restrict__ src,
                           bf16_t* __restrict__ dst, int ldd, int coloff,
                           long long n) {
    long long i = (long long)blockIdx.x * blockDim.x + threadIdx.x;
    if (i >= n) return;
    int m = (int)(i >> 8), c = (int)(i & 255);
    dst[(size_t)m * ldd + coloff + c] = (bf16_t)src[i];
}

// ---------------------------------------------------------------------------
// Edge scatter: one wave per edge, S[ety][dst][:] += X[src][:]
__global__ __launch_bounds__(256)
void k_scatter_f32(const float* __restrict__ X, const int* __restrict__ src,
                   const int* __restrict__ dst, const int* __restrict__ ety,
                   float* __restrict__ S, int E, int ndst) {
    int lane = threadIdx.x & 31;
    int e = blockIdx.x * 8 + (threadIdx.x >> 5);
    if (e >= E) return;
    int s = src[e], d = dst[e], r = ety[e];
    const float* xs = X + (size_t)s * 256;
    float* op = S + ((size_t)r * ndst + d) * 256;
#pragma unroll
    for (int i = 0; i < 8; ++i)
        atomicAdd(op + lane + i * 32, xs[lane + i * 32]);
}

__global__ __launch_bounds__(256)
void k_scatter_bf16(const bf16_t* __restrict__ X, const int* __restrict__ src,
                    const int* __restrict__ dst, const int* __restrict__ ety,
                    float* __restrict__ S, int E, int ndst) {
    int lane = threadIdx.x & 31;
    int e = blockIdx.x * 8 + (threadIdx.x >> 5);
    if (e >= E) return;
    int s = src[e], d = dst[e], r = ety[e];
    const bf16_t* xs = X + (size_t)s * 256;
    float* op = S + ((size_t)r * ndst + d) * 256;
#pragma unroll
    for (int i = 0; i < 8; ++i)
        atomicAdd(op + lane + i * 32, (float)xs[lane + i * 32]);
}

// ---------------------------------------------------------------------------
// bf16 WMMA GEMM: C[M,ldc] = A[M,K] (bf16, row-major) * BT[N,K]^T (bf16) + bias
// Block: 256 threads = 8 waves. Wave w -> 32x64 tile: rows [by*256+32w, +32),
// cols [bx*64, +64). Two A fragments reuse the same four B fragments ->
// 8 WMMAs per 12 b128 loads per K-step. M rows beyond the end are clamp-
// loaded (valid memory) and store-guarded; EXEC stays all-ones for WMMA.
__global__ __launch_bounds__(256)
void k_gemm_bf16(const bf16_t* __restrict__ A, const bf16_t* __restrict__ BT,
                 const float* __restrict__ bias, float* __restrict__ C,
                 int M, int K, int Nout, int ldc) {
    const int lane = threadIdx.x & 31;
    const int wave = threadIdx.x >> 5;
    const int m0 = blockIdx.y * 256 + wave * 32;
    const int n0 = blockIdx.x * 64;
    if (m0 >= M) return;           // wave-uniform exit
    const bool has2 = (m0 + 16) < M;

    v8f zero8 = {0,0,0,0,0,0,0,0};
    v8f acc[8];
#pragma unroll
    for (int i = 0; i < 8; ++i) acc[i] = zero8;

    const int hi  = lane >> 4;     // 0 | 1
    const int akb = hi << 4;       // A K-phase: 0 | 16
    const int bkb = hi << 4;       // B K-phase: 0 | 16
    const int arow0 = m0 + (lane & 15);
    int arow1 = m0 + 16 + (lane & 15);
    if (arow1 >= M) arow1 = M - 1; // padding rows: computed, never stored
    const bf16_t* ap0 = A + (size_t)arow0 * K + akb;
    const bf16_t* ap1 = A + (size_t)arow1 * K + akb;
    const bf16_t* bp  = BT + (size_t)(n0 + (lane & 15)) * K + bkb;

    for (int k0 = 0; k0 < K; k0 += 32) {
        __builtin_prefetch((const void*)(ap0 + k0 + 512), 0, 3);
        v16bf a0 = *(const v16bf*)(ap0 + k0);
        v16bf a1 = *(const v16bf*)(ap1 + k0);
        v16bf b0 = *(const v16bf*)(bp + k0);
        v16bf b1 = *(const v16bf*)(bp + (size_t)16 * K + k0);
        v16bf b2 = *(const v16bf*)(bp + (size_t)32 * K + k0);
        v16bf b3 = *(const v16bf*)(bp + (size_t)48 * K + k0);
        acc[0] = __builtin_amdgcn_wmma_f32_16x16x32_bf16(false, a0, false, b0,
                                                         (short)0, acc[0], false, false);
        acc[1] = __builtin_amdgcn_wmma_f32_16x16x32_bf16(false, a0, false, b1,
                                                         (short)0, acc[1], false, false);
        acc[2] = __builtin_amdgcn_wmma_f32_16x16x32_bf16(false, a0, false, b2,
                                                         (short)0, acc[2], false, false);
        acc[3] = __builtin_amdgcn_wmma_f32_16x16x32_bf16(false, a0, false, b3,
                                                         (short)0, acc[3], false, false);
        acc[4] = __builtin_amdgcn_wmma_f32_16x16x32_bf16(false, a1, false, b0,
                                                         (short)0, acc[4], false, false);
        acc[5] = __builtin_amdgcn_wmma_f32_16x16x32_bf16(false, a1, false, b1,
                                                         (short)0, acc[5], false, false);
        acc[6] = __builtin_amdgcn_wmma_f32_16x16x32_bf16(false, a1, false, b2,
                                                         (short)0, acc[6], false, false);
        acc[7] = __builtin_amdgcn_wmma_f32_16x16x32_bf16(false, a1, false, b3,
                                                         (short)0, acc[7], false, false);
    }

    // C/D layout: VGPR g -> row base + g + 8*hi, col n0 + (lane&15) + 16*t
    const int crow0 = m0 + (hi << 3);
    const int crow1 = m0 + 16 + (hi << 3);
    const int col0  = n0 + (lane & 15);
#pragma unroll
    for (int t = 0; t < 4; ++t) {
        int c = col0 + t * 16;
        if (c < Nout) {
            float bv = bias ? bias[c] : 0.f;
            float* cp0 = C + (size_t)crow0 * ldc + c;
            v8f av0 = acc[t];
#pragma unroll
            for (int g = 0; g < 8; ++g)
                cp0[(size_t)g * ldc] = av0[g] + bv;
            if (has2) {
                float* cp1 = C + (size_t)crow1 * ldc + c;
                v8f av1 = acc[4 + t];
#pragma unroll
                for (int g = 0; g < 8; ++g)
                    cp1[(size_t)g * ldc] = av1[g] + bv;
            }
        }
    }
}

// ---------------------------------------------------------------------------
// Per-column sum / sum-of-squares over [M,256] (batch dim reduction)
__global__ __launch_bounds__(256)
void k_colstats(const float* __restrict__ H, int M, float* __restrict__ stats) {
    int c = threadIdx.x;
    int r0 = blockIdx.x * 256;
    int r1 = r0 + 256; if (r1 > M) r1 = M;
    float s = 0.f, s2 = 0.f;
    for (int r = r0; r < r1; ++r) {
        float v = H[(size_t)r * 256 + c];
        s += v; s2 += v * v;
    }
    atomicAdd(&stats[c], s);
    atomicAdd(&stats[256 + c], s2);
}

__global__ void k_bnfinal(const float* __restrict__ stats,
                          const float* __restrict__ gamma,
                          const float* __restrict__ beta,
                          float invM, float* __restrict__ ss) {
    int c = threadIdx.x;
    float mu  = stats[c] * invM;
    float var = stats[256 + c] * invM - mu * mu;
    float sc  = gamma[c] * rsqrtf(var + 1e-5f);
    ss[c] = sc;
    ss[256 + c] = beta[c] - mu * sc;
}

// BN apply + ELU/ReLU + f32->bf16 (optionally duplicate first rows2 rows
// into a second column-slice destination, e.g. next layer's self-loop block)
__global__ void k_bnact(const float* __restrict__ H, const float* __restrict__ ss,
                        bf16_t* __restrict__ out1, int ld1,
                        bf16_t* __restrict__ out2, int ld2, int rows2,
                        long long n, int relu) {
    long long i = (long long)blockIdx.x * blockDim.x + threadIdx.x;
    if (i >= n) return;
    int r = (int)(i >> 8), c = (int)(i & 255);
    float v = H[i] * ss[c] + ss[256 + c];
    v = relu ? fmaxf(v, 0.f) : (v > 0.f ? v : (expf(v) - 1.f));
    bf16_t bv = (bf16_t)v;
    out1[(size_t)r * ld1 + c] = bv;
    if (out2 && r < rows2) out2[(size_t)r * ld2 + c] = bv;
}

// ---------------------------------------------------------------------------
extern "C" void kernel_launch(void* const* d_in, const int* in_sizes, int n_in,
                              void* d_out, int out_size, void* d_ws, size_t ws_size,
                              hipStream_t stream) {
    (void)n_in; (void)out_size; (void)ws_size;
    const float* x   = (const float*)d_in[0];
    const int* src0  = (const int*)d_in[1];
    const int* dst0  = (const int*)d_in[2];
    const int* ety0  = (const int*)d_in[3];
    const int* src1  = (const int*)d_in[4];
    const int* dst1  = (const int*)d_in[5];
    const int* ety1  = (const int*)d_in[6];
    const float* W0  = (const float*)d_in[9];
    const float* Wl0 = (const float*)d_in[10];
    const float* b0  = (const float*)d_in[11];
    const float* g0  = (const float*)d_in[12];
    const float* be0 = (const float*)d_in[13];
    const float* W1  = (const float*)d_in[14];
    const float* Wl1 = (const float*)d_in[15];
    const float* b1  = (const float*)d_in[16];
    const float* g1  = (const float*)d_in[17];
    const float* be1 = (const float*)d_in[18];
    const float* Wm1 = (const float*)d_in[19];
    const float* bm1 = (const float*)d_in[20];
    const float* gm  = (const float*)d_in[21];
    const float* bem = (const float*)d_in[22];
    const float* Wm2 = (const float*)d_in[23];
    const float* bm2 = (const float*)d_in[24];

    const int E0 = in_sizes[1];
    const int E1 = in_sizes[4];
    const int ND0 = 50000, ND1 = 10000;     // n_dst0 / n_dst1 (static)
    const int K1 = 1536;                    // (5 relations + self-loop) * 256

    // ---- workspace layout (bytes); layer-1 buffers alias dead S0 region ----
    char* ws = (char*)d_ws;
    float*  S0   = (float*)(ws + 0);              // 5*50000*256 f32 = 256,000,000
    float*  S1   = (float*)(ws + 0);              //  51,200,000  (alias, after S0 dead)
    bf16_t* A1   = (bf16_t*)(ws + 51200000);      //  30,720,000
    float*  H1   = (float*)(ws + 81920000);       //  10,240,000
    bf16_t* H1a  = (bf16_t*)(ws + 92160000);      //   5,120,000
    float*  M1   = (float*)(ws + 97280000);       //  10,240,000
    bf16_t* M1a  = (bf16_t*)(ws + 107520000);     //   5,120,000
    bf16_t* A0   = (bf16_t*)(ws + 256000000);     // 153,600,000
    bf16_t* H0a  = (bf16_t*)(ws + 256000000);     //  25,600,000 (alias A0, after GEMM0)
    float*  H0   = (float*)(ws + 409600000);      //  51,200,000
    bf16_t* WT0  = (bf16_t*)(ws + 460800000);     //     786,432
    bf16_t* WT1  = (bf16_t*)(ws + 461586432);     //     786,432
    bf16_t* WmT1 = (bf16_t*)(ws + 462372864);     //     131,072
    bf16_t* WmT2 = (bf16_t*)(ws + 462503936);     //     131,072
    float*  stats= (float*)(ws + 462635008);      //       2,048
    float*  ssb  = (float*)(ws + 462637056);      //       2,048

    auto nblk = [](long long n, int tpb) { return (unsigned)((n + tpb - 1) / tpb); };

    // ---- weight prep: transposed bf16 [N,K] layouts ----
    for (int r = 0; r < 5; ++r)
        k_wtrans<<<256, 256, 0, stream>>>(W0 + (size_t)r * 65536, 256, WT0, K1, r * 256, 256);
    k_wtrans<<<256, 256, 0, stream>>>(Wl0, 256, WT0, K1, 1280, 256);
    for (int r = 0; r < 5; ++r)
        k_wtrans<<<256, 256, 0, stream>>>(W1 + (size_t)r * 65536, 256, WT1, K1, r * 256, 256);
    k_wtrans<<<256, 256, 0, stream>>>(Wl1, 256, WT1, K1, 1280, 256);
    k_wtrans<<<256, 256, 0, stream>>>(Wm1, 256, WmT1, 256, 0, 256);
    k_wtrans<<<256, 256, 0, stream>>>(Wm2, 153, WmT2, 256, 0, 256);  // zero-padded cols

    // ---- layer 0: scatter-aggregate, GEMM, BN+ELU ----
    k_zero<<<nblk(64000000LL, 256), 256, 0, stream>>>(S0, 64000000LL);
    k_scatter_f32<<<(E0 + 7) / 8, 256, 0, stream>>>(x, src0, dst0, ety0, S0, E0, ND0);
    for (int r = 0; r < 5; ++r)
        k_cvt_cols<<<nblk(12800000LL, 256), 256, 0, stream>>>(
            S0 + (size_t)r * ND0 * 256, A0, K1, r * 256, 12800000LL);
    k_cvt_cols<<<nblk(12800000LL, 256), 256, 0, stream>>>(x, A0, K1, 1280, 12800000LL);
    {
        dim3 g(4, (ND0 + 255) / 256);
        k_gemm_bf16<<<g, 256, 0, stream>>>(A0, WT0, b0, H0, ND0, K1, 256, 256);
    }
    k_zero<<<2, 256, 0, stream>>>(stats, 512LL);
    k_colstats<<<(ND0 + 255) / 256, 256, 0, stream>>>(H0, ND0, stats);
    k_bnfinal<<<1, 256, 0, stream>>>(stats, g0, be0, 1.f / (float)ND0, ssb);
    k_bnact<<<nblk(12800000LL, 256), 256, 0, stream>>>(
        H0, ssb, H0a, 256, A1 + 1280, K1, ND1, 12800000LL, 0);

    // ---- layer 1 ----
    k_zero<<<nblk(12800000LL, 256), 256, 0, stream>>>(S1, 12800000LL);
    k_scatter_bf16<<<(E1 + 7) / 8, 256, 0, stream>>>(H0a, src1, dst1, ety1, S1, E1, ND1);
    for (int r = 0; r < 5; ++r)
        k_cvt_cols<<<nblk(2560000LL, 256), 256, 0, stream>>>(
            S1 + (size_t)r * ND1 * 256, A1, K1, r * 256, 2560000LL);
    {
        dim3 g(4, (ND1 + 255) / 256);
        k_gemm_bf16<<<g, 256, 0, stream>>>(A1, WT1, b1, H1, ND1, K1, 256, 256);
    }
    k_zero<<<2, 256, 0, stream>>>(stats, 512LL);
    k_colstats<<<(ND1 + 255) / 256, 256, 0, stream>>>(H1, ND1, stats);
    k_bnfinal<<<1, 256, 0, stream>>>(stats, g1, be1, 1.f / (float)ND1, ssb);
    k_bnact<<<nblk(2560000LL, 256), 256, 0, stream>>>(
        H1, ssb, H1a, 256, (bf16_t*)nullptr, 0, 0, 2560000LL, 0);

    // ---- MLP head ----
    {
        dim3 g(4, (ND1 + 255) / 256);
        k_gemm_bf16<<<g, 256, 0, stream>>>(H1a, WmT1, bm1, M1, ND1, 256, 256, 256);
    }
    k_zero<<<2, 256, 0, stream>>>(stats, 512LL);
    k_colstats<<<(ND1 + 255) / 256, 256, 0, stream>>>(M1, ND1, stats);
    k_bnfinal<<<1, 256, 0, stream>>>(stats, gm, bem, 1.f / (float)ND1, ssb);
    k_bnact<<<nblk(2560000LL, 256), 256, 0, stream>>>(
        M1, ssb, M1a, 256, (bf16_t*)nullptr, 0, 0, 2560000LL, 1);
    {
        dim3 g(3, (ND1 + 255) / 256);   // 3*64 = 192 >= 153 computed cols
        k_gemm_bf16<<<g, 256, 0, stream>>>(M1a, WmT2, bm2, (float*)d_out,
                                           ND1, 256, 153, 153);
    }
}
